// MultiheadAttention_13932873909171
// MI455X (gfx1250) — compile-verified, matrix operand order checked by reading, not run on hardware
//
#include <hip/hip_runtime.h>

#define AK 1024   // attention size
#define LK 4096   // sequence length
#define HK 16     // heads
#define HSK 64    // head size
#define MK 64     // landmarks
#define BHK 64    // B*H
#define ROWSK 16384  // B*L

typedef __attribute__((ext_vector_type(16))) __bf16 v16bf;
typedef __attribute__((ext_vector_type(8)))  __bf16 v8bf;
typedef __attribute__((ext_vector_type(8)))  float  v8f;

// ---------------------------------------------------------------- WMMA core
__device__ __forceinline__ v8f wmma_bf16(v16bf a, v16bf b, v8f c) {
  // D = A(16x32 bf16) * B(32x16 bf16) + C(16x16 f32)
  return __builtin_amdgcn_wmma_f32_16x16x32_bf16(false, a, false, b, (short)0, c,
                                                 false, false);
}

// A/B operand loader from row-major bf16 (A: rows of A; B: rows of B^T).
// ISA 16-bit operand layout: lane L -> row L%16; elements e<8 -> K = 8g+e,
// e>=8 -> K = 16+8g+(e-8), g = L/16.  Two contiguous 16B runs per lane.
__device__ __forceinline__ v16bf ldop_bf16(const __bf16* base, int ld, int lane) {
  int r = lane & 15, g = lane >> 4;
  const __bf16* p = base + (size_t)r * ld + 8 * g;
  v16bf o;
#pragma unroll
  for (int e = 0; e < 8; ++e) { o[e] = p[e]; o[e + 8] = p[16 + e]; }
  return o;
}

// Same layout but source is fp32 (convert on the fly).
__device__ __forceinline__ v16bf ldop_f32(const float* base, int ld, int lane) {
  int r = lane & 15, g = lane >> 4;
  const float* p = base + (size_t)r * ld + 8 * g;
  v16bf o;
#pragma unroll
  for (int e = 0; e < 8; ++e) { o[e] = (__bf16)p[e]; o[e + 8] = (__bf16)p[16 + e]; }
  return o;
}

// Store a 16x16 f32 D tile (lane holds column n = lane%16, rows m = r + 8g).
__device__ __forceinline__ void st_tile_f32(float* dst, int ld, int m0, int n0,
                                            int lane, v8f c) {
  int n = n0 + (lane & 15), g = lane >> 4;
#pragma unroll
  for (int r = 0; r < 8; ++r) dst[(size_t)(m0 + 8 * g + r) * ld + n] = c[r];
}

// --------- operand loaders from fp32 LDS (64x64 matrices, pinv chain) -------
__device__ __forceinline__ v16bf ldA_lds(const float* Am, int m0, int k0, int lane) {
  int r = m0 + (lane & 15), g = lane >> 4;
  v16bf o;
#pragma unroll
  for (int e = 0; e < 16; ++e) {
    int kk = k0 + 8 * g + (e & 7) + ((e >= 8) ? 16 : 0);
    o[e] = (__bf16)Am[r * 64 + kk];
  }
  return o;
}
// B operand with optional elementwise transform  f(B)[k][n] = diag*I - B
__device__ __forceinline__ v16bf ldB_lds(const float* Bm, int k0, int n0, int lane,
                                         float diag, int dosub) {
  int n = n0 + (lane & 15), g = lane >> 4;
  v16bf o;
#pragma unroll
  for (int e = 0; e < 16; ++e) {
    int kk = k0 + 8 * g + (e & 7) + ((e >= 8) ? 16 : 0);
    float v = Bm[kk * 64 + n];
    if (dosub) v = ((kk == n) ? diag : 0.0f) - v;
    o[e] = (__bf16)v;
  }
  return o;
}

// D = A @ f(B), 64x64x64, 8 waves (2 subtiles each), result scaled by sc.
__device__ __forceinline__ void mm64(const float* Am, const float* Bm, float* Dm,
                                     int tid, float diag, int dosub, float sc) {
  int lane = tid & 31, w = tid >> 5, sm = w & 3, snp = w >> 2;
  int m0 = 16 * sm, n0 = 32 * snp, n1 = n0 + 16;
  v8f c0 = {}, c1 = {};
#pragma unroll
  for (int k0 = 0; k0 < 64; k0 += 32) {
    v16bf a = ldA_lds(Am, m0, k0, lane);
    c0 = wmma_bf16(a, ldB_lds(Bm, k0, n0, lane, diag, dosub), c0);
    c1 = wmma_bf16(a, ldB_lds(Bm, k0, n1, lane, diag, dosub), c1);
  }
  int n = lane & 15, g = lane >> 4;
#pragma unroll
  for (int r = 0; r < 8; ++r) {
    Dm[(m0 + 8 * g + r) * 64 + n0 + n] = c0[r] * sc;
    Dm[(m0 + 8 * g + r) * 64 + n1 + n] = c1[r] * sc;
  }
}

// ============================ kernels =======================================

// Transpose + convert a 1024x1024 fp32 weight to bf16 [n][k].
__global__ __launch_bounds__(256) void wprep_kernel(const float* __restrict__ W,
                                                    __bf16* __restrict__ Wt) {
  int idx = blockIdx.x * 256 + threadIdx.x;
  int n = idx >> 10, kk = idx & 1023;
  Wt[idx] = (__bf16)W[(size_t)kk * AK + n];
}

// C = X(fp32) @ W  via  Wt[n][k] bf16.  Block tile 128x64, each of 8 waves
// computes a 16x64 strip: 4 WMMA accumulators share one A operand -> 4x A
// reuse in registers.  Head-split bf16 output (+optional transposed copy for
// V), optional scale (q: 1/sqrt(HS)).
__global__ __launch_bounds__(256) void proj_gemm_kernel(const float* __restrict__ X,
                                                        const __bf16* __restrict__ Wt,
                                                        __bf16* Out, __bf16* OutT,
                                                        float scale) {
  int row0 = blockIdx.x * 128, col0 = blockIdx.y * 64;
  int tid = threadIdx.x, lane = tid & 31, w = tid >> 5;  // w = subtile row 0..7
  const float* Ab = X + (size_t)(row0 + 16 * w) * AK;
  const __bf16* Bb = Wt + (size_t)col0 * AK;
  v8f acc[4] = {{}, {}, {}, {}};
#pragma unroll 2
  for (int k0 = 0; k0 < AK; k0 += 32) {
    if (k0 + 256 < AK) {
      __builtin_prefetch(Ab + k0 + 256, 0, 1);
      __builtin_prefetch(Bb + k0 + 256, 0, 1);
    }
    v16bf a = ldop_f32(Ab + k0, AK, lane);
#pragma unroll
    for (int j = 0; j < 4; ++j)
      acc[j] = wmma_bf16(a, ldop_bf16(Bb + (size_t)(16 * j) * AK + k0, AK, lane),
                         acc[j]);
  }
  int n = lane & 15, g = lane >> 4;
  int h = col0 >> 6;                       // col0 is head-aligned
  int gr0 = row0 + 16 * w + 8 * g;
  int b = gr0 >> 12, l0 = gr0 & (LK - 1);  // 8 rows stay in one batch
  if (Out) {
#pragma unroll
    for (int j = 0; j < 4; ++j) {
      int hs = 16 * j + n;
      __bf16* dst = Out + (((size_t)(b * HK + h)) * LK + l0) * HSK + hs;
#pragma unroll
      for (int r = 0; r < 8; ++r) dst[(size_t)r * HSK] = (__bf16)(acc[j][r] * scale);
    }
  }
  if (OutT) {  // V^T [bh][hs][l]: 8 contiguous l per lane -> single b128 store
#pragma unroll
    for (int j = 0; j < 4; ++j) {
      int hs = 16 * j + n;
      v8bf t;
#pragma unroll
      for (int r = 0; r < 8; ++r) t[r] = (__bf16)(acc[j][r] * scale);
      *(v8bf*)(OutT + ((size_t)(b * HK + h) * HSK + hs) * LK + l0) = t;
    }
  }
}

// Landmarks: mean over 64-element segments -> [bh][M][HS] bf16.
__global__ __launch_bounds__(256) void landmark_kernel(const __bf16* __restrict__ S,
                                                       __bf16* __restrict__ D) {
  int bh = blockIdx.x;
  for (int idx = threadIdx.x; idx < MK * HSK; idx += 256) {
    int m = idx >> 6, hs = idx & 63;
    const __bf16* p = S + ((size_t)bh * LK + m * 64) * HSK + hs;
    float s = 0.f;
#pragma unroll 8
    for (int d = 0; d < 64; ++d) s += (float)p[(size_t)d * HSK];
    D[(size_t)bh * MK * HSK + idx] = (__bf16)(s * (1.0f / 64.0f));
  }
}

// kv = softmax(q_l @ k^T) @ v   (flash-style online softmax over L), per (b,h).
// Output transposed kv^T [bh][HS][M] bf16 (contiguous m per lane).
__global__ __launch_bounds__(256) void s3v_kernel(const __bf16* __restrict__ ql,
                                                  const __bf16* __restrict__ kmat,
                                                  const __bf16* __restrict__ vt,
                                                  __bf16* __restrict__ kvt) {
  __shared__ float Sf[64 * 64];
  __shared__ __bf16 Pb[64 * 64];
  __shared__ float rmax[64], rsum[64], corr[64];
  int bh = blockIdx.x;
  int tid = threadIdx.x, lane = tid & 31, w = tid >> 5, sm = w & 3, snp = w >> 2;
  const __bf16* Aq = ql + (size_t)bh * MK * HSK + 16 * sm * HSK;
  v16bf a0 = ldop_bf16(Aq, HSK, lane);         // q_l operands are loop-invariant
  v16bf a1 = ldop_bf16(Aq + 32, HSK, lane);
  v8f o0 = {}, o1 = {};
  if (tid < 64) { rmax[tid] = -1e30f; rsum[tid] = 0.f; }
  __syncthreads();
  for (int c = 0; c < LK; c += 64) {
    const __bf16* Kb0 = kmat + ((size_t)bh * LK + c + 32 * snp) * HSK;
    const __bf16* Kb1 = Kb0 + 16 * HSK;
    v8f s0 = {}, s1 = {};
    s0 = wmma_bf16(a0, ldop_bf16(Kb0, HSK, lane), s0);
    s0 = wmma_bf16(a1, ldop_bf16(Kb0 + 32, HSK, lane), s0);
    s1 = wmma_bf16(a0, ldop_bf16(Kb1, HSK, lane), s1);
    s1 = wmma_bf16(a1, ldop_bf16(Kb1 + 32, HSK, lane), s1);
    st_tile_f32(Sf, 64, 16 * sm, 32 * snp, lane, s0);
    st_tile_f32(Sf, 64, 16 * sm, 32 * snp + 16, lane, s1);
    __syncthreads();
    if (tid < 64) {  // online softmax, one row per thread
      float mo = rmax[tid], mx = mo;
      for (int j = 0; j < 64; ++j) mx = fmaxf(mx, Sf[tid * 64 + j]);
      float cr = __expf(mo - mx);
      float s = rsum[tid] * cr;
      for (int j = 0; j < 64; ++j) {
        float e = __expf(Sf[tid * 64 + j] - mx);
        s += e;
        Pb[tid * 64 + j] = (__bf16)e;
      }
      rmax[tid] = mx; rsum[tid] = s; corr[tid] = cr;
    }
    __syncthreads();
    int g = lane >> 4;
#pragma unroll
    for (int r = 0; r < 8; ++r) {  // rescale running accumulators
      float cc = corr[16 * sm + 8 * g + r];
      o0[r] *= cc; o1[r] *= cc;
    }
    const __bf16* Pa = Pb + 16 * sm * 64;
    v16bf p0 = ldop_bf16(Pa, 64, lane);
    v16bf p1 = ldop_bf16(Pa + 32, 64, lane);
    const __bf16* Vb0 = vt + ((size_t)bh * HSK + 32 * snp) * LK + c;
    const __bf16* Vb1 = Vb0 + (size_t)16 * LK;
    o0 = wmma_bf16(p0, ldop_bf16(Vb0, LK, lane), o0);
    o0 = wmma_bf16(p1, ldop_bf16(Vb0 + 32, LK, lane), o0);
    o1 = wmma_bf16(p0, ldop_bf16(Vb1, LK, lane), o1);
    o1 = wmma_bf16(p1, ldop_bf16(Vb1 + 32, LK, lane), o1);
    __syncthreads();
  }
  int n = lane & 15, g = lane >> 4;
  v8bf t0, t1;
#pragma unroll
  for (int r = 0; r < 8; ++r) {
    float inv = 1.0f / rsum[16 * sm + 8 * g + r];
    t0[r] = (__bf16)(o0[r] * inv);
    t1[r] = (__bf16)(o1[r] * inv);
  }
  int hs0 = 32 * snp + n, hs1 = hs0 + 16;
  *(v8bf*)(kvt + ((size_t)bh * HSK + hs0) * MK + 16 * sm + 8 * g) = t0;
  *(v8bf*)(kvt + ((size_t)bh * HSK + hs1) * MK + 16 * sm + 8 * g) = t1;
}

// s2 = softmax(q_l @ k_l^T); z* = Moore-Penrose(s2, 6 iters); z_kv = z* @ kv.
// Entire chain in LDS (5 x 16KB fp32 buffers), one workgroup per (b,h).
__global__ __launch_bounds__(256) void pinv_kernel(const __bf16* __restrict__ ql,
                                                   const __bf16* __restrict__ kl,
                                                   const __bf16* __restrict__ kvt,
                                                   __bf16* __restrict__ zkvt) {
  extern __shared__ float dsm[];
  float* Xf = dsm;              // s2 (x)
  float* Zf = dsm + 4096;       // z
  float* Q0 = dsm + 8192;       // xz
  float* Q1 = dsm + 12288;
  float* Q2 = dsm + 16384;
  float* csum = dsm + 20480;    // 64
  float* rsum = dsm + 20544;    // 64
  float* alpha = dsm + 20608;   // 1
  int bh = blockIdx.x;
  int tid = threadIdx.x, lane = tid & 31, w = tid >> 5, sm = w & 3, snp = w >> 2;
  // scores q_l @ k_l^T -> Xf
  const __bf16* Aq = ql + (size_t)bh * MK * HSK + 16 * sm * HSK;
  const __bf16* B0 = kl + (size_t)bh * MK * HSK + 32 * snp * HSK;
  const __bf16* B1 = B0 + 16 * HSK;
  v8f s0 = {}, s1 = {};
#pragma unroll
  for (int k0 = 0; k0 < HSK; k0 += 32) {
    v16bf a = ldop_bf16(Aq + k0, HSK, lane);
    s0 = wmma_bf16(a, ldop_bf16(B0 + k0, HSK, lane), s0);
    s1 = wmma_bf16(a, ldop_bf16(B1 + k0, HSK, lane), s1);
  }
  st_tile_f32(Xf, 64, 16 * sm, 32 * snp, lane, s0);
  st_tile_f32(Xf, 64, 16 * sm, 32 * snp + 16, lane, s1);
  __syncthreads();
  if (tid < 64) {  // softmax row, in place
    float mx = -1e30f;
    for (int j = 0; j < 64; ++j) mx = fmaxf(mx, Xf[tid * 64 + j]);
    float s = 0.f;
    for (int j = 0; j < 64; ++j) { float e = __expf(Xf[tid * 64 + j] - mx); s += e; Xf[tid * 64 + j] = e; }
    float inv = 1.0f / s, rs = 0.f;
    for (int j = 0; j < 64; ++j) { float v = Xf[tid * 64 + j] * inv; Xf[tid * 64 + j] = v; rs += v; }
    rsum[tid] = rs;
  }
  __syncthreads();
  if (tid < 64) {
    float cs = 0.f;
    for (int i = 0; i < 64; ++i) cs += Xf[i * 64 + tid];
    csum[tid] = cs;
  }
  __syncthreads();
  if (tid == 0) {
    float mc = 0.f, mr = 0.f;
    for (int i = 0; i < 64; ++i) { mc = fmaxf(mc, csum[i]); mr = fmaxf(mr, rsum[i]); }
    alpha[0] = 1.0f / (mc * mr);
  }
  __syncthreads();
  float inva = alpha[0];
  for (int idx = tid; idx < 4096; idx += 256) {  // z0 = x^T / (col*row)
    int m = idx >> 6, nn = idx & 63;
    Zf[idx] = Xf[nn * 64 + m] * inva;
  }
  __syncthreads();
  for (int it = 0; it < 6; ++it) {
    mm64(Xf, Zf, Q0, tid, 0.f, 0, 1.0f);  __syncthreads();   // xz
    mm64(Q0, Q0, Q1, tid, 7.0f, 1, 1.0f); __syncthreads();   // xz(7I-xz)
    mm64(Q0, Q1, Q2, tid, 15.0f, 1, 1.0f); __syncthreads();  // xz(15I-..)
    mm64(Zf, Q2, Q1, tid, 13.0f, 1, 0.25f); __syncthreads(); // .25 z(13I-..)
    for (int idx = tid; idx < 4096; idx += 256) Zf[idx] = Q1[idx];
    __syncthreads();
  }
  // z_kv = z* @ kv  -> store transposed [bh][HS][M]
  const __bf16* KV0 = kvt + (size_t)bh * HSK * MK + 32 * snp * MK;
  const __bf16* KV1 = KV0 + 16 * MK;
  v8f o0 = {}, o1 = {};
#pragma unroll
  for (int k0 = 0; k0 < MK; k0 += 32) {
    v16bf a = ldA_lds(Zf, 16 * sm, k0, lane);
    o0 = wmma_bf16(a, ldop_bf16(KV0 + k0, MK, lane), o0);
    o1 = wmma_bf16(a, ldop_bf16(KV1 + k0, MK, lane), o1);
  }
  int n = lane & 15, g = lane >> 4;
  v8bf t0, t1;
#pragma unroll
  for (int r = 0; r < 8; ++r) { t0[r] = (__bf16)o0[r]; t1[r] = (__bf16)o1[r]; }
  int hs0 = 32 * snp + n, hs1 = hs0 + 16;
  *(v8bf*)(zkvt + ((size_t)bh * HSK + hs0) * MK + 16 * sm + 8 * g) = t0;
  *(v8bf*)(zkvt + ((size_t)bh * HSK + hs1) * MK + 16 * sm + 8 * g) = t1;
}

// out_heads = softmax(q @ k_l^T) @ z_kv, merged-head bf16 [rows][A].
__global__ __launch_bounds__(256) void s1out_kernel(const __bf16* __restrict__ qb,
                                                    const __bf16* __restrict__ klb,
                                                    const __bf16* __restrict__ zkvt,
                                                    __bf16* __restrict__ outh) {
  __shared__ float Sf[64 * 64];
  __shared__ __bf16 Pb[64 * 64];
  int bh = blockIdx.x;
  int l0 = blockIdx.y * 64;
  int tid = threadIdx.x, lane = tid & 31, w = tid >> 5, sm = w & 3, snp = w >> 2;
  const __bf16* Aq = qb + ((size_t)bh * LK + l0 + 16 * sm) * HSK;
  const __bf16* B0 = klb + (size_t)bh * MK * HSK + 32 * snp * HSK;
  const __bf16* B1 = B0 + 16 * HSK;
  v8f s0 = {}, s1 = {};
#pragma unroll
  for (int k0 = 0; k0 < HSK; k0 += 32) {
    v16bf a = ldop_bf16(Aq + k0, HSK, lane);
    s0 = wmma_bf16(a, ldop_bf16(B0 + k0, HSK, lane), s0);
    s1 = wmma_bf16(a, ldop_bf16(B1 + k0, HSK, lane), s1);
  }
  st_tile_f32(Sf, 64, 16 * sm, 32 * snp, lane, s0);
  st_tile_f32(Sf, 64, 16 * sm, 32 * snp + 16, lane, s1);
  __syncthreads();
  if (tid < 64) {  // full softmax, row length = M = 64
    float mx = -1e30f;
    for (int j = 0; j < 64; ++j) mx = fmaxf(mx, Sf[tid * 64 + j]);
    float s = 0.f;
    float e[64];
    for (int j = 0; j < 64; ++j) { e[j] = __expf(Sf[tid * 64 + j] - mx); s += e[j]; }
    float inv = 1.0f / s;
    for (int j = 0; j < 64; ++j) Pb[tid * 64 + j] = (__bf16)(e[j] * inv);
  }
  __syncthreads();
  const __bf16* Pa = Pb + 16 * sm * 64;
  const __bf16* Z0 = zkvt + (size_t)bh * HSK * MK + 32 * snp * MK;
  const __bf16* Z1 = Z0 + 16 * MK;
  v8f o0 = {}, o1 = {};
#pragma unroll
  for (int k0 = 0; k0 < MK; k0 += 32) {
    v16bf p = ldop_bf16(Pa + k0, 64, lane);
    o0 = wmma_bf16(p, ldop_bf16(Z0 + k0, MK, lane), o0);
    o1 = wmma_bf16(p, ldop_bf16(Z1 + k0, MK, lane), o1);
  }
  int b = bh >> 4, h = bh & 15, n = lane & 15, g = lane >> 4;
  int hs0 = 32 * snp + n, hs1 = hs0 + 16;
#pragma unroll
  for (int r = 0; r < 8; ++r) {
    size_t gr = (size_t)b * LK + l0 + 16 * sm + 8 * g + r;
    outh[gr * AK + h * HSK + hs0] = (__bf16)o0[r];
    outh[gr * AK + h * HSK + hs1] = (__bf16)o1[r];
  }
}

// out = out_heads @ Wo + bo  (fp32 output).  Same 128x64 / 16x64-per-wave
// register blocking as proj_gemm.
__global__ __launch_bounds__(256) void final_gemm_kernel(const __bf16* __restrict__ Xh,
                                                         const __bf16* __restrict__ Wt,
                                                         const float* __restrict__ bias,
                                                         float* __restrict__ Out) {
  int row0 = blockIdx.x * 128, col0 = blockIdx.y * 64;
  int tid = threadIdx.x, lane = tid & 31, w = tid >> 5;
  const __bf16* Ab = Xh + (size_t)(row0 + 16 * w) * AK;
  const __bf16* Bb = Wt + (size_t)col0 * AK;
  v8f acc[4] = {{}, {}, {}, {}};
#pragma unroll 2
  for (int k0 = 0; k0 < AK; k0 += 32) {
    if (k0 + 256 < AK) {
      __builtin_prefetch(Ab + k0 + 256, 0, 1);
      __builtin_prefetch(Bb + k0 + 256, 0, 1);
    }
    v16bf a = ldop_bf16(Ab + k0, AK, lane);
#pragma unroll
    for (int j = 0; j < 4; ++j)
      acc[j] = wmma_bf16(a, ldop_bf16(Bb + (size_t)(16 * j) * AK + k0, AK, lane),
                         acc[j]);
  }
  int n = lane & 15, g = lane >> 4;
  int gr0 = row0 + 16 * w + 8 * g;
#pragma unroll
  for (int j = 0; j < 4; ++j) {
    int gc = col0 + 16 * j + n;
    float bv = bias[gc];
#pragma unroll
    for (int r = 0; r < 8; ++r)
      Out[(size_t)(gr0 + r) * AK + gc] = acc[j][r] + bv;
  }
}

// ============================ host glue =====================================
extern "C" void kernel_launch(void* const* d_in, const int* in_sizes, int n_in,
                              void* d_out, int out_size, void* d_ws, size_t ws_size,
                              hipStream_t stream) {
  (void)in_sizes; (void)n_in; (void)out_size; (void)ws_size;
  const float* x  = (const float*)d_in[0];
  const float* Wk = (const float*)d_in[1];
  const float* Wq = (const float*)d_in[2];
  const float* Wv = (const float*)d_in[3];
  const float* Wo = (const float*)d_in[4];
  const float* bo = (const float*)d_in[5];
  float* out = (float*)d_out;

  char* ws = (char*)d_ws;
  const size_t SZ_W   = (size_t)AK * AK * sizeof(__bf16);        // 2 MB
  const size_t SZ_QKV = (size_t)BHK * LK * HSK * sizeof(__bf16); // 32 MB
  const size_t SZ_LM  = (size_t)BHK * MK * HSK * sizeof(__bf16); // 512 KB
  size_t off = 0;
  __bf16* Wqt  = (__bf16*)(ws + off); off += SZ_W;
  __bf16* Wkt  = (__bf16*)(ws + off); off += SZ_W;
  __bf16* Wvt  = (__bf16*)(ws + off); off += SZ_W;
  __bf16* Wot  = (__bf16*)(ws + off); off += SZ_W;
  __bf16* qb   = (__bf16*)(ws + off); off += SZ_QKV;
  __bf16* kb   = (__bf16*)(ws + off); off += SZ_QKV;
  __bf16* vtb  = (__bf16*)(ws + off); off += SZ_QKV;
  __bf16* qlb  = (__bf16*)(ws + off); off += SZ_LM;
  __bf16* klb  = (__bf16*)(ws + off); off += SZ_LM;
  __bf16* kvtb = (__bf16*)(ws + off); off += SZ_LM;
  __bf16* zkvtb= (__bf16*)(ws + off); off += SZ_LM;
  __bf16* outhb= (__bf16*)(ws + off); off += SZ_QKV;

  // 1. weight convert + transpose
  wprep_kernel<<<AK * AK / 256, 256, 0, stream>>>(Wq, Wqt);
  wprep_kernel<<<AK * AK / 256, 256, 0, stream>>>(Wk, Wkt);
  wprep_kernel<<<AK * AK / 256, 256, 0, stream>>>(Wv, Wvt);
  wprep_kernel<<<AK * AK / 256, 256, 0, stream>>>(Wo, Wot);

  // 2. projections (q scaled by 1/sqrt(HS)=1/8); v stored transposed only
  dim3 gg(ROWSK / 128, AK / 64);
  proj_gemm_kernel<<<gg, 256, 0, stream>>>(x, Wqt, qb, nullptr, 0.125f);
  proj_gemm_kernel<<<gg, 256, 0, stream>>>(x, Wkt, kb, nullptr, 1.0f);
  proj_gemm_kernel<<<gg, 256, 0, stream>>>(x, Wvt, nullptr, vtb, 1.0f);

  // 3. landmarks
  landmark_kernel<<<BHK, 256, 0, stream>>>(qb, qlb);
  landmark_kernel<<<BHK, 256, 0, stream>>>(kb, klb);

  // 4. kv^T = (softmax(q_l k^T) v)^T  (flash over L)
  s3v_kernel<<<BHK, 256, 0, stream>>>(qlb, kb, vtb, kvtb);

  // 5. pseudo-inverse chain + z_kv^T
  pinv_kernel<<<BHK, 256, 20736 * sizeof(float), stream>>>(qlb, klb, kvtb, zkvtb);

  // 6. out_heads = softmax(q k_l^T) z_kv
  s1out_kernel<<<dim3(BHK, LK / 64), 256, 0, stream>>>(qb, klb, zkvtb, outhb);

  // 7. final projection + bias
  final_gemm_kernel<<<gg, 256, 0, stream>>>(outhb, Wot, bo, out);
}
